// MultiHeadAttention_89352499626036
// MI455X (gfx1250) — matrix operand order repaired and checked by reading, not compiled, not run on hardware
//
#include <hip/hip_runtime.h>

#define D_MODEL 1024
#define NHEADS  16
#define DK      64
#define BATCH   2
#define SEQ     2048
#define ROWS    (BATCH*SEQ)

#define APITCH 40   // halfwords: GEMM A tiles (32 cols), 80B row pitch
#define BPITCH 72   // halfwords: GEMM B tiles (64 cols), 144B row pitch
#define KVB    64   // flash kv block width
#define KPITCH 72   // flash K/V/P tile pitch (64 cols + pad), 144B rows

typedef __attribute__((ext_vector_type(16))) __bf16 v16bf;
typedef __attribute__((ext_vector_type(8)))  float  v8f;

struct __align__(16) U4  { unsigned int x, y, z, w; };
struct __align__(16) F4  { float x, y, z, w; };
struct __align__(8)  US4 { unsigned short a, b, c, d; };

union Frag { v16bf v; U4 q[2]; };

// Hardware bf16 convert (v_cvt_pk_bf16_f32 via backend).
__device__ __forceinline__ unsigned short f2bf(float f) {
  union { __bf16 h; unsigned short u; } cv;
  cv.h = (__bf16)f;
  return cv.u;
}

__device__ __forceinline__ v8f wmma_bf16(v16bf a, v16bf b, v8f c) {
  return __builtin_amdgcn_wmma_f32_16x16x32_bf16(
      false, a, false, b, (short)0, c, false, false);
}

// --- VALU 16-lane xor-max reduction via v_permlane16_b32 (no LDS traffic) --
__device__ __forceinline__ float pl16(float x, int s0, int s1) {
  int v = __float_as_int(x);
  int r = __builtin_amdgcn_permlane16(v, v, s0, s1, false, false);
  return __int_as_float(r);
}
__device__ __forceinline__ float rowmax16(float v) {
  v = fmaxf(v, pl16(v, 0x67452301, 0xEFCDAB89));   // xor 1
  v = fmaxf(v, pl16(v, 0x54761032, 0xDCFE98BA));   // xor 2
  v = fmaxf(v, pl16(v, 0x32107654, 0xBA98FEDC));   // xor 4
  v = fmaxf(v, pl16(v, 0xFEDCBA98, 0x76543210));   // xor 8
  return v;
}

// --- CDNA5 async global->LDS copy (ASYNCcnt-tracked, bypasses VGPRs) -------
__device__ __forceinline__ unsigned lds_off32(const void* p) {
  return (unsigned)(unsigned long long)(uintptr_t)p;
}
__device__ __forceinline__ void async_ld_b128(unsigned lds_byte, const void* g) {
  asm volatile("global_load_async_to_lds_b128 %0, %1, off"
               :: "v"(lds_byte), "v"((unsigned long long)(uintptr_t)g)
               : "memory");
}
__device__ __forceinline__ void wait_async0() {
  asm volatile("s_wait_asynccnt 0x0" ::: "memory");
}

// A fragment (16x32 bf16, MxK): lane&15 = row M; K along VGPRs.
__device__ __forceinline__ v16bf load_afrag(const unsigned short* p, int pitch) {
  int lane = threadIdx.x & 31;
  const unsigned short* r = p + (lane & 15) * pitch + ((lane >> 4) << 3);
  Frag f;
  f.q[0] = *(const U4*)(r);
  f.q[1] = *(const U4*)(r + 16);
  return f.v;
}

// B fragment (32x16 bf16, KxN): lane&15 = col N; K along VGPRs (same order as A).
__device__ __forceinline__ v16bf load_bfrag(const unsigned short* p, int pitch) {
  int lane = threadIdx.x & 31;
  const unsigned short* r = p + (lane & 15) + ((lane >> 4) << 3) * pitch;
  union { v16bf v; unsigned short u[16]; } f;
  #pragma unroll
  for (int s = 0; s < 8; ++s) {
    f.u[s] = r[s * pitch];
    f.u[s + 8] = r[(16 + s) * pitch];
  }
  return f.v;
}

// ---------------------------------------------------------------------------
// K0: weights fp32 [N][K] -> bf16 transposed [K][N]
// ---------------------------------------------------------------------------
__global__ __launch_bounds__(256) void wconv_k(
    const float* __restrict__ w0, const float* __restrict__ w1,
    const float* __restrict__ w2, const float* __restrict__ w3,
    unsigned short* __restrict__ o0, unsigned short* __restrict__ o1,
    unsigned short* __restrict__ o2, unsigned short* __restrict__ o3) {
  const float* w; unsigned short* o;
  switch (blockIdx.z) {
    case 0:  w = w0; o = o0; break;
    case 1:  w = w1; o = o1; break;
    case 2:  w = w2; o = o2; break;
    default: w = w3; o = o3; break;
  }
  __shared__ float tile[32][33];
  int tx = threadIdx.x & 31, ty = threadIdx.x >> 5;
  int n0 = blockIdx.y * 32, k0 = blockIdx.x * 32;
  #pragma unroll
  for (int r = 0; r < 32; r += 8)
    tile[ty + r][tx] = w[(size_t)(n0 + ty + r) * D_MODEL + k0 + tx];
  __syncthreads();
  #pragma unroll
  for (int r = 0; r < 32; r += 8)
    o[(size_t)(k0 + ty + r) * D_MODEL + n0 + tx] = f2bf(tile[tx][ty + r]);
}

// A-tile staging helpers for proj (fp32 -> bf16 through VGPRs)
__device__ __forceinline__ void projA_load(const float* xbase, int t, int kb, F4* xa) {
  #pragma unroll
  for (int j = 0; j < 4; ++j) {
    int id = t + 256 * j;
    int row = id >> 3, cb = (id & 7) * 4;
    xa[j] = *(const F4*)(xbase + (size_t)row * D_MODEL + kb + cb);
  }
}
__device__ __forceinline__ void projA_store(unsigned short* Abuf, int t, const F4* xa) {
  #pragma unroll
  for (int j = 0; j < 4; ++j) {
    int id = t + 256 * j;
    int row = id >> 3, cb = (id & 7) * 4;
    US4 h; h.a = f2bf(xa[j].x); h.b = f2bf(xa[j].y);
    h.c = f2bf(xa[j].z); h.d = f2bf(xa[j].w);
    *(US4*)(Abuf + row * APITCH + cb) = h;
  }
}

// ---------------------------------------------------------------------------
// K1: Y = X(fp32) @ Wt(bf16) + bias; scatter to head layouts in bf16.
// 128x64 WG tile, 32x32 per wave, double-buffered LDS, async B copies.
// ---------------------------------------------------------------------------
template <int MODE>
__global__ __launch_bounds__(256) void proj_gemm(
    const float* __restrict__ X, const unsigned short* __restrict__ Wt,
    const float* __restrict__ bias, unsigned short* __restrict__ out) {
  __shared__ __align__(16) unsigned short As[2][128 * APITCH];
  __shared__ __align__(16) unsigned short Bs[2][32 * BPITCH];
  int t = threadIdx.x, lane = t & 31, wv = t >> 5;
  int rb = blockIdx.x * 128, nb = blockIdx.y * 64;
  int r = wv & 3, c = wv >> 2;
  v8f acc[2][2];
  #pragma unroll
  for (int i = 0; i < 2; ++i)
    #pragma unroll
    for (int j = 0; j < 2; ++j) acc[i][j] = (v8f){0,0,0,0,0,0,0,0};

  int brow = t >> 3, bcb = (t & 7) * 8;
  unsigned bs_off[2] = { lds_off32(Bs[0] + brow * BPITCH + bcb),
                         lds_off32(Bs[1] + brow * BPITCH + bcb) };
  const unsigned short* bsrc = Wt + (size_t)brow * D_MODEL + nb + bcb;
  const float* xbase = X + (size_t)rb * D_MODEL;

  { // prologue: stage k-step 0 into buffer 0
    async_ld_b128(bs_off[0], bsrc);
    F4 xa[4];
    projA_load(xbase, t, 0, xa);
    projA_store(As[0], t, xa);
  }

  const int NS = D_MODEL / 32;
  for (int ks = 0; ks < NS; ++ks) {
    int buf = ks & 1;
    wait_async0();
    __syncthreads();
    F4 xa[4];
    bool next = (ks + 1 < NS);
    if (next) {
      async_ld_b128(bs_off[buf ^ 1], bsrc + (size_t)(ks + 1) * 32 * D_MODEL);
      projA_load(xbase, t, (ks + 1) * 32, xa);
    }
    v16bf a0 = load_afrag(As[buf] + (32 * r) * APITCH, APITCH);
    v16bf a1 = load_afrag(As[buf] + (32 * r + 16) * APITCH, APITCH);
    v16bf b0 = load_bfrag(Bs[buf] + (2 * c + 0) * 16, BPITCH);
    v16bf b1 = load_bfrag(Bs[buf] + (2 * c + 1) * 16, BPITCH);
    acc[0][0] = wmma_bf16(a0, b0, acc[0][0]);
    acc[0][1] = wmma_bf16(a0, b1, acc[0][1]);
    acc[1][0] = wmma_bf16(a1, b0, acc[1][0]);
    acc[1][1] = wmma_bf16(a1, b1, acc[1][1]);
    if (next) projA_store(As[buf ^ 1], t, xa);
  }

  const float QSCALE = 0.125f * 1.44269504088896f;
  #pragma unroll
  for (int sub = 0; sub < 2; ++sub) {
    #pragma unroll
    for (int cc = 0; cc < 2; ++cc) {
      int gn = nb + 32 * c + 16 * cc + (lane & 15);
      float bval = bias[gn];
      int h = gn >> 6, d = gn & 63;
      #pragma unroll
      for (int i = 0; i < 8; ++i) {
        int gm = rb + 32 * r + 16 * sub + i + ((lane >> 4) << 3);
        int bidx = gm >> 11, s = gm & (SEQ - 1);
        float val = acc[sub][cc][i] + bval;
        if (MODE == 0) {
          size_t addr = ((size_t)(bidx * NHEADS + h) * SEQ + s) * DK + d;
          out[addr] = f2bf(val * QSCALE);
        } else if (MODE == 1) {
          size_t addr = ((size_t)(bidx * NHEADS + h) * DK + d) * SEQ + s;
          out[addr] = f2bf(val);
        } else {
          size_t addr = ((size_t)(bidx * NHEADS + h) * SEQ + s) * DK + d;
          out[addr] = f2bf(val);
        }
      }
    }
  }
}

// ---------------------------------------------------------------------------
// K2: flash attention, 64-wide KV blocks, double-buffered async K/V tiles.
// Row sums via WMMA against an all-ones B fragment; row max via
// v_permlane16-based VALU reduction (LDS pipe stays free for fragments).
// ---------------------------------------------------------------------------
__global__ __launch_bounds__(256) void flash_attn(
    const unsigned short* __restrict__ Q, const unsigned short* __restrict__ Kt,
    const unsigned short* __restrict__ V, unsigned short* __restrict__ Ctx) {
  __shared__ __align__(16) unsigned short Ks[2][64 * KPITCH];   // [d=64][kv=64]
  __shared__ __align__(16) unsigned short Vs[2][64 * KPITCH];   // [kv=64][d=64]
  __shared__ __align__(16) unsigned short Ps[8 * 16 * KPITCH];  // per-wave P
  int t = threadIdx.x, lane = t & 31, wv = t >> 5;
  int bh = blockIdx.y;
  int q0 = blockIdx.x * 128;
  unsigned short* Psw = Ps + wv * 16 * KPITCH;

  // Register-resident Q A-fragments (d-steps 0 and 1), A layout from global.
  int kb2 = (lane >> 4) << 3;
  const unsigned short* qrow =
      Q + ((size_t)bh * SEQ + q0 + 16 * wv + (lane & 15)) * DK;
  Frag qa0, qa1;
  qa0.q[0] = *(const U4*)(qrow + kb2);
  qa0.q[1] = *(const U4*)(qrow + kb2 + 16);
  qa1.q[0] = *(const U4*)(qrow + 32 + kb2);
  qa1.q[1] = *(const U4*)(qrow + 32 + kb2 + 16);

  // All-ones bf16 B fragment (1.0 = 0x3F80) for WMMA row sums.
  Frag onef;
  {
    U4 o1 = {0x3F803F80u, 0x3F803F80u, 0x3F803F80u, 0x3F803F80u};
    onef.q[0] = o1; onef.q[1] = o1;
  }

  v8f acc[4], accl;
  #pragma unroll
  for (int cc = 0; cc < 4; ++cc) acc[cc] = (v8f){0,0,0,0,0,0,0,0};
  accl = (v8f){0,0,0,0,0,0,0,0};
  float mi[8];
  #pragma unroll
  for (int i = 0; i < 8; ++i) mi[i] = -1e30f;

  // Staging: each thread copies two 16B chunks per tile (rows r0 and r0+32).
  int r0 = t >> 3, cb = (t & 7) * 8;
  unsigned ks_off[2][2], vs_off[2][2];
  #pragma unroll
  for (int b = 0; b < 2; ++b) {
    ks_off[b][0] = lds_off32(Ks[b] + r0 * KPITCH + cb);
    ks_off[b][1] = lds_off32(Ks[b] + (r0 + 32) * KPITCH + cb);
    vs_off[b][0] = lds_off32(Vs[b] + r0 * KPITCH + cb);
    vs_off[b][1] = lds_off32(Vs[b] + (r0 + 32) * KPITCH + cb);
  }
  const unsigned short* ksrc0 = Kt + ((size_t)bh * DK + r0) * SEQ + cb;
  const unsigned short* ksrc1 = Kt + ((size_t)bh * DK + r0 + 32) * SEQ + cb;
  const unsigned short* vsrc  = V + ((size_t)bh * SEQ) * DK + cb;

  // prologue: stage block 0 into buffer 0
  async_ld_b128(ks_off[0][0], ksrc0);
  async_ld_b128(ks_off[0][1], ksrc1);
  async_ld_b128(vs_off[0][0], vsrc + (size_t)r0 * DK);
  async_ld_b128(vs_off[0][1], vsrc + (size_t)(r0 + 32) * DK);

  const int NB = SEQ / KVB;
  for (int jb = 0; jb < NB; ++jb) {
    int buf = jb & 1;
    wait_async0();
    __syncthreads();
    if (jb + 1 < NB) {                        // prefetch next KV block
      int j1 = (jb + 1) * KVB;
      async_ld_b128(ks_off[buf ^ 1][0], ksrc0 + j1);
      async_ld_b128(ks_off[buf ^ 1][1], ksrc1 + j1);
      async_ld_b128(vs_off[buf ^ 1][0], vsrc + (size_t)(j1 + r0) * DK);
      async_ld_b128(vs_off[buf ^ 1][1], vsrc + (size_t)(j1 + r0 + 32) * DK);
    }

    // scores (16q x 64kv), log2 domain (Q pre-scaled by 0.125*log2e)
    v8f s[4];
    #pragma unroll
    for (int cc = 0; cc < 4; ++cc) s[cc] = (v8f){0,0,0,0,0,0,0,0};
    #pragma unroll
    for (int dstep = 0; dstep < 2; ++dstep) {
      v16bf a = dstep ? qa1.v : qa0.v;
      #pragma unroll
      for (int cc = 0; cc < 4; ++cc) {
        v16bf b = load_bfrag(Ks[buf] + dstep * 32 * KPITCH + cc * 16, KPITCH);
        s[cc] = wmma_bf16(a, b, s[cc]);
      }
    }

    // online softmax: row max via 4 permlane16 steps (pure VALU).
    float al[8];
    #pragma unroll
    for (int i = 0; i < 8; ++i) {
      float vm = fmaxf(fmaxf(s[0][i], s[1][i]), fmaxf(s[2][i], s[3][i]));
      vm = rowmax16(vm);
      float mn = fmaxf(mi[i], vm);
      al[i] = exp2f(mi[i] - mn);
      mi[i] = mn;
      int prow = i + ((lane >> 4) << 3);        // C-layout -> LDS scratch
      #pragma unroll
      for (int cc = 0; cc < 4; ++cc)
        Psw[prow * KPITCH + cc * 16 + (lane & 15)] = f2bf(exp2f(s[cc][i] - mn));
    }
    #pragma unroll
    for (int i = 0; i < 8; ++i) {
      acc[0][i] *= al[i]; acc[1][i] *= al[i];
      acc[2][i] *= al[i]; acc[3][i] *= al[i];
      accl[i]   *= al[i];
    }

    // P(16x64) @ V(64x64) + row-sum accumulator
    v16bf pa0 = load_afrag(Psw + 0,  KPITCH);
    v16bf pa1 = load_afrag(Psw + 32, KPITCH);
    accl = wmma_bf16(pa0, onef.v, accl);
    accl = wmma_bf16(pa1, onef.v, accl);
    #pragma unroll
    for (int cc = 0; cc < 4; ++cc) {
      v16bf b0 = load_bfrag(Vs[buf] + 0  * KPITCH + cc * 16, KPITCH);
      v16bf b1 = load_bfrag(Vs[buf] + 32 * KPITCH + cc * 16, KPITCH);
      acc[cc] = wmma_bf16(pa0, b0, acc[cc]);
      acc[cc] = wmma_bf16(pa1, b1, acc[cc]);
    }
  }

  int b = bh >> 4, h = bh & 15;
  #pragma unroll
  for (int i = 0; i < 8; ++i) {
    float l = accl[i];
    float inv = l > 0.f ? 1.f / l : 0.f;
    int srow = q0 + 16 * wv + i + ((lane >> 4) << 3);
    unsigned short* dst =
        Ctx + ((size_t)(b * SEQ + srow)) * D_MODEL + h * DK + (lane & 15);
    #pragma unroll
    for (int cc = 0; cc < 4; ++cc)
      dst[cc * 16] = f2bf(acc[cc][i] * inv);
  }
}

// ---------------------------------------------------------------------------
// K3: out = Ctx(bf16) @ Wo_t(bf16) + bo -> fp32.  Double-buffered, all-async.
// ---------------------------------------------------------------------------
__global__ __launch_bounds__(256) void out_gemm(
    const unsigned short* __restrict__ Ctx, const unsigned short* __restrict__ Wt,
    const float* __restrict__ bias, float* __restrict__ out) {
  __shared__ __align__(16) unsigned short As[2][128 * APITCH];
  __shared__ __align__(16) unsigned short Bs[2][32 * BPITCH];
  int t = threadIdx.x, lane = t & 31, wv = t >> 5;
  int rb = blockIdx.x * 128, nb = blockIdx.y * 64;
  int r = wv & 3, c = wv >> 2;
  v8f acc[2][2];
  #pragma unroll
  for (int i = 0; i < 2; ++i)
    #pragma unroll
    for (int j = 0; j < 2; ++j) acc[i][j] = (v8f){0,0,0,0,0,0,0,0};

  int arow0 = t >> 2, acb0 = (t & 3) * 8;
  int arow1 = (t + 256) >> 2, acb1 = ((t + 256) & 3) * 8;
  int brow = t >> 3, bcb = (t & 7) * 8;
  unsigned as_off0[2] = { lds_off32(As[0] + arow0 * APITCH + acb0),
                          lds_off32(As[1] + arow0 * APITCH + acb0) };
  unsigned as_off1[2] = { lds_off32(As[0] + arow1 * APITCH + acb1),
                          lds_off32(As[1] + arow1 * APITCH + acb1) };
  unsigned bs_off[2]  = { lds_off32(Bs[0] + brow * BPITCH + bcb),
                          lds_off32(Bs[1] + brow * BPITCH + bcb) };
  const unsigned short* asrc0 = Ctx + (size_t)(rb + arow0) * D_MODEL + acb0;
  const unsigned short* asrc1 = Ctx + (size_t)(rb + arow1) * D_MODEL + acb1;
  const unsigned short* bsrc  = Wt + (size_t)brow * D_MODEL + nb + bcb;

  async_ld_b128(as_off0[0], asrc0);
  async_ld_b128(as_off1[0], asrc1);
  async_ld_b128(bs_off[0], bsrc);

  const int NS = D_MODEL / 32;
  for (int ks = 0; ks < NS; ++ks) {
    int buf = ks & 1;
    wait_async0();
    __syncthreads();
    if (ks + 1 < NS) {
      int kb = (ks + 1) * 32;
      async_ld_b128(as_off0[buf ^ 1], asrc0 + kb);
      async_ld_b128(as_off1[buf ^ 1], asrc1 + kb);
      async_ld_b128(bs_off[buf ^ 1], bsrc + (size_t)kb * D_MODEL);
    }
    v16bf a0 = load_afrag(As[buf] + (32 * r) * APITCH, APITCH);
    v16bf a1 = load_afrag(As[buf] + (32 * r + 16) * APITCH, APITCH);
    v16bf b0 = load_bfrag(Bs[buf] + (2 * c + 0) * 16, BPITCH);
    v16bf b1 = load_bfrag(Bs[buf] + (2 * c + 1) * 16, BPITCH);
    acc[0][0] = wmma_bf16(a0, b0, acc[0][0]);
    acc[0][1] = wmma_bf16(a0, b1, acc[0][1]);
    acc[1][0] = wmma_bf16(a1, b0, acc[1][0]);
    acc[1][1] = wmma_bf16(a1, b1, acc[1][1]);
  }
  #pragma unroll
  for (int sub = 0; sub < 2; ++sub) {
    #pragma unroll
    for (int cc = 0; cc < 2; ++cc) {
      int gn = nb + 32 * c + 16 * cc + (lane & 15);
      float bval = bias[gn];
      #pragma unroll
      for (int i = 0; i < 8; ++i) {
        int gm = rb + 32 * r + 16 * sub + i + ((lane >> 4) << 3);
        out[(size_t)gm * D_MODEL + gn] = acc[sub][cc][i] + bval;
      }
    }
  }
}

extern "C" void kernel_launch(void* const* d_in, const int* in_sizes, int n_in,
                              void* d_out, int out_size, void* d_ws, size_t ws_size,
                              hipStream_t stream) {
  (void)in_sizes; (void)n_in; (void)out_size;
  const float* q  = (const float*)d_in[0];
  const float* k  = (const float*)d_in[1];
  const float* v  = (const float*)d_in[2];
  const float* wq = (const float*)d_in[3];
  const float* bq = (const float*)d_in[4];
  const float* wk = (const float*)d_in[5];
  const float* bk = (const float*)d_in[6];
  const float* wv = (const float*)d_in[7];
  const float* bv = (const float*)d_in[8];
  const float* wo = (const float*)d_in[9];
  const float* bo = (const float*)d_in[10];

  unsigned short* ws = (unsigned short*)d_ws;
  const size_t W = (size_t)D_MODEL * D_MODEL;   // 1M elems per weight
  const size_t T = (size_t)ROWS * D_MODEL;      // 4M elems per activation
  if (ws_size < (4 * W + 4 * T) * sizeof(unsigned short)) return;  // ~40MB

  unsigned short* Wqt = ws;
  unsigned short* Wkt = ws + W;
  unsigned short* Wvt = ws + 2 * W;
  unsigned short* Wot = ws + 3 * W;
  unsigned short* Qs  = ws + 4 * W;
  unsigned short* Kts = Qs + T;
  unsigned short* Vvs = Kts + T;
  unsigned short* Cx  = Vvs + T;

  wconv_k<<<dim3(32, 32, 4), 256, 0, stream>>>(wq, wk, wv, wo, Wqt, Wkt, Wvt, Wot);
  proj_gemm<0><<<dim3(ROWS / 128, D_MODEL / 64), 256, 0, stream>>>(q, Wqt, bq, Qs);
  proj_gemm<1><<<dim3(ROWS / 128, D_MODEL / 64), 256, 0, stream>>>(k, Wkt, bk, Kts);
  proj_gemm<2><<<dim3(ROWS / 128, D_MODEL / 64), 256, 0, stream>>>(v, Wvt, bv, Vvs);
  flash_attn<<<dim3(SEQ / 128, BATCH * NHEADS), 256, 0, stream>>>(Qs, Kts, Vvs, Cx);
  out_gemm<<<dim3(ROWS / 128, D_MODEL / 64), 256, 0, stream>>>(Cx, Wot, bo, (float*)d_out);
}
